// myRNN_4595615007334
// MI455X (gfx1250) — matrix operand order repaired and checked by reading, not compiled, not run on hardware
//
#include <hip/hip_runtime.h>
#include <stdint.h>
#include <stddef.h>

// Problem constants (match reference)
#define B_ 64
#define S_ 512
#define I_ 512
#define H_ 1024

typedef __bf16 bf16;
typedef bf16  v16bf __attribute__((ext_vector_type(16)));
typedef bf16  v8bf  __attribute__((ext_vector_type(8)));
typedef float v8f   __attribute__((ext_vector_type(8)));

struct GruParams {
  const bf16 *wr0, *wz0, *wc0;      // layer0 gate weights, bf16, (H, I+H) row-major
  const bf16 *wr1, *wz1, *wc1;      // layer1 gate weights, bf16, (H, 2H) row-major
  const bf16 *xb;                   // x in bf16, (B, S, I)
  float *h0, *h1;                   // hidden states fp32 (B, H)
  bf16 *h0b0, *h0b1;                // double-buffered bf16 mirror of h0
  bf16 *h1b;                        // bf16 mirror of h1
  bf16 *hr0b, *hr1b;                // bf16 h*r buffers
  float *z0, *z1;                   // z gate values fp32
  unsigned *barCnt, *barGen;        // grid barrier state
  const float *br0, *bz0, *bc0;
  const float *br1, *bz1, *bc1;
  const float *wreg, *breg;         // regressor (O,H), (O)
  float *out;                       // (B, O) fp32
};

// ---------------- WMMA fragment loaders (CDNA5 16x16x32 bf16 layouts) ----------

// A fragment: 16(M) x 32(K), activations row-major with row stride `stride`.
// Lane L: m = L&15, hi = L>>4. Lane holds K = 16*hi + {0..15}
// -> one contiguous 32-byte span per lane (loaded as two 16-byte chunks).
__device__ __forceinline__ v16bf load_a(const bf16* __restrict__ base,
                                        size_t stride, int lane) {
  const int m  = lane & 15;
  const int hi = lane >> 4;
  const bf16* p = base + (size_t)m * stride + hi * 16;
  v8bf lo = *(const v8bf*)(p);
  v8bf hv = *(const v8bf*)(p + 8);
  return __builtin_shufflevector(lo, hv, 0,1,2,3,4,5,6,7,8,9,10,11,12,13,14,15);
}

// B fragment: 32(K) x 16(N). B = W^T, W stored (N rows, K contiguous) row-major,
// so column n of B is row n of W. Lane L: n = L&15, hi = L>>4; lane holds
// K = 16*hi .. 16*hi+15 -> one contiguous 32-byte chunk of the W row.
__device__ __forceinline__ v16bf load_b(const bf16* __restrict__ base,
                                        size_t K, int lane) {
  const int n  = lane & 15;
  const int hi = lane >> 4;
  return *(const v16bf*)(base + (size_t)n * K + hi * 16);
}

__device__ __forceinline__ v8f wmma_bf16(v16bf a, v16bf b, v8f c) {
  return __builtin_amdgcn_wmma_f32_16x16x32_bf16(
      /*neg_a=*/false, a, /*neg_b=*/false, b,
      /*c_mod=*/(short)0, c, /*reuse_a=*/false, /*reuse_b=*/false);
}

// One 16x16 output tile of D = A * W^T, where A's K-range is the concatenation
// of two row-major sources: a1 covers k in [0,len1), a2 covers [len1,len1+len2).
// `w` points at W + (tile_col0 * K); K = len1 + len2.
__device__ __forceinline__ v8f gemm_tile(const bf16* __restrict__ a1, size_t s1, int len1,
                                         const bf16* __restrict__ a2, size_t s2, int len2,
                                         const bf16* __restrict__ w, size_t K, int lane) {
  v8f acc = {0.f, 0.f, 0.f, 0.f, 0.f, 0.f, 0.f, 0.f};
#pragma unroll 4
  for (int k = 0; k < len1; k += 32)
    acc = wmma_bf16(load_a(a1 + k, s1, lane), load_b(w + k, K, lane), acc);
#pragma unroll 4
  for (int k = 0; k < len2; k += 32)
    acc = wmma_bf16(load_a(a2 + k, s2, lane), load_b(w + len1 + k, K, lane), acc);
  return acc;
}

// ---------------- grid-wide barrier (persistent kernel) ------------------------

__device__ __forceinline__ void grid_barrier(unsigned* cnt, unsigned* gen,
                                             unsigned nBlocks, unsigned& localGen) {
  __threadfence();       // publish this thread's phase writes device-wide
  __syncthreads();
  if (threadIdx.x == 0) {
    const unsigned target = localGen + 1;
    const unsigned prev =
        __hip_atomic_fetch_add(cnt, 1u, __ATOMIC_ACQ_REL, __HIP_MEMORY_SCOPE_AGENT);
    if (prev == nBlocks - 1) {
      __hip_atomic_store(cnt, 0u, __ATOMIC_RELAXED, __HIP_MEMORY_SCOPE_AGENT);
      __hip_atomic_fetch_add(gen, 1u, __ATOMIC_ACQ_REL, __HIP_MEMORY_SCOPE_AGENT);
    } else {
      while (__hip_atomic_load(gen, __ATOMIC_ACQUIRE, __HIP_MEMORY_SCOPE_AGENT) < target) {
        __builtin_amdgcn_s_sleep(2);   // short yield: barrier wake latency matters (1024 barriers)
      }
    }
    localGen = target;
  }
  __syncthreads();
}

__device__ __forceinline__ float sigmoid_f(float x) {
  return 1.f / (1.f + __expf(-x));
}

// ---------------- persistent, software-pipelined GRU kernel --------------------
//
// Dependency graph per step t:  A(t)->B(t)->A(t+1)   (layer 0)
//                               C(t)->D(t)->C(t+1)   (layer 1)
//                               B(t)->C(t)           (cross edge)
// => run G1 = A(t) || C(t-1), barrier, G2 = B(t) || D(t-1), barrier.
// h0b is double-buffered: B(t) writes buf[t&1]; A(t+1), C(t), D(t) read buf[t&1].

__global__ void __launch_bounds__(256)
gru_persistent(GruParams p) {
  const int lane        = threadIdx.x & 31;
  const int wavesPerBlk = blockDim.x >> 5;
  const int waveId      = blockIdx.x * wavesPerBlk + (threadIdx.x >> 5);
  const int totalWaves  = gridDim.x * wavesPerBlk;
  const unsigned nBlocks = gridDim.x;
  unsigned gen = 0;

  const size_t xRowStride = (size_t)S_ * I_;  // x is (B, S, I): batch row stride

  for (int tt = 0; tt <= S_; ++tt) {
    const int tA = tt;        // layer-0 step (A/B); valid while tA < S_
    const int tC = tt - 1;    // layer-1 step (C/D); valid while tC >= 0
    // h0b(tA-1) lives in buf[(tA-1)&1]; B(tA) writes buf[tA&1].
    const bf16* h0b_prev = (tA & 1) ? p.h0b0 : p.h0b1;
    bf16*       h0b_new  = (tA & 1) ? p.h0b1 : p.h0b0;
    const bf16* xrow = p.xb + (size_t)tA * I_;

    // ==== G1: A(tA) gates-0 (tiles 0..511)  ||  C(tC) gates-1 (tiles 512..1023)
    for (int tile = waveId; tile < 1024; tile += totalWaves) {
      if (tile < 512) {
        if (tA >= S_) continue;
        const int gate = tile >> 8;
        const int rem  = tile & 255;
        const int nt   = rem >> 2;
        const int mt   = rem & 3;
        const bf16*  Wg   = gate ? p.wz0 : p.wr0;
        const float* bias = gate ? p.bz0 : p.br0;
        v8f acc = gemm_tile(xrow + (size_t)(mt * 16) * xRowStride, xRowStride, I_,
                            h0b_prev + (size_t)(mt * 16) * H_, H_, H_,
                            Wg + (size_t)(nt * 16) * (I_ + H_), (size_t)(I_ + H_), lane);
        const int n = lane & 15, hi = lane >> 4;
        const int col = nt * 16 + n, row0 = mt * 16 + hi * 8;
        const float bv = bias[col];
#pragma unroll
        for (int e = 0; e < 8; ++e) {
          const size_t idx = (size_t)(row0 + e) * H_ + col;
          const float s = sigmoid_f(acc[e] + bv);
          if (gate == 0) p.hr0b[idx] = (bf16)(p.h0[idx] * s);  // h0 * r
          else           p.z0[idx] = s;
        }
      } else {
        if (tC < 0) continue;
        const int t2   = tile - 512;
        const int gate = t2 >> 8;
        const int rem  = t2 & 255;
        const int nt   = rem >> 2;
        const int mt   = rem & 3;
        const bf16*  Wg   = gate ? p.wz1 : p.wr1;
        const float* bias = gate ? p.bz1 : p.br1;
        v8f acc = gemm_tile(h0b_prev + (size_t)(mt * 16) * H_, H_, H_,   // h0b(tC)
                            p.h1b + (size_t)(mt * 16) * H_, H_, H_,      // h1b(tC-1)
                            Wg + (size_t)(nt * 16) * (2 * H_), (size_t)(2 * H_), lane);
        const int n = lane & 15, hi = lane >> 4;
        const int col = nt * 16 + n, row0 = mt * 16 + hi * 8;
        const float bv = bias[col];
#pragma unroll
        for (int e = 0; e < 8; ++e) {
          const size_t idx = (size_t)(row0 + e) * H_ + col;
          const float s = sigmoid_f(acc[e] + bv);
          if (gate == 0) p.hr1b[idx] = (bf16)(p.h1[idx] * s);  // h1 * r1
          else           p.z1[idx] = s;
        }
      }
    }
    grid_barrier(p.barCnt, p.barGen, nBlocks, gen);

    // ==== G2: B(tA) cand-0 + h0 update (tiles 0..255) || D(tC) cand-1 + h1 update
    for (int tile = waveId; tile < 512; tile += totalWaves) {
      if (tile < 256) {
        if (tA >= S_) continue;
        const int nt = tile >> 2, mt = tile & 3;
        v8f acc = gemm_tile(xrow + (size_t)(mt * 16) * xRowStride, xRowStride, I_,
                            p.hr0b + (size_t)(mt * 16) * H_, H_, H_,
                            p.wc0 + (size_t)(nt * 16) * (I_ + H_), (size_t)(I_ + H_), lane);
        const int n = lane & 15, hi = lane >> 4;
        const int col = nt * 16 + n, row0 = mt * 16 + hi * 8;
        const float bv = p.bc0[col];
#pragma unroll
        for (int e = 0; e < 8; ++e) {
          const size_t idx = (size_t)(row0 + e) * H_ + col;
          const float can = tanhf(acc[e] + bv);
          const float z   = p.z0[idx];
          const float hn  = p.h0[idx] * z + (1.f - z) * can;   // h*z + (1-z)*can
          p.h0[idx]   = hn;
          h0b_new[idx] = (bf16)hn;
        }
      } else {
        if (tC < 0) continue;
        const int t2 = tile - 256;
        const int nt = t2 >> 2, mt = t2 & 3;
        v8f acc = gemm_tile(h0b_prev + (size_t)(mt * 16) * H_, H_, H_,   // h0b(tC)
                            p.hr1b + (size_t)(mt * 16) * H_, H_, H_,
                            p.wc1 + (size_t)(nt * 16) * (2 * H_), (size_t)(2 * H_), lane);
        const int n = lane & 15, hi = lane >> 4;
        const int col = nt * 16 + n, row0 = mt * 16 + hi * 8;
        const float bv = p.bc1[col];
#pragma unroll
        for (int e = 0; e < 8; ++e) {
          const size_t idx = (size_t)(row0 + e) * H_ + col;
          const float can = tanhf(acc[e] + bv);
          const float z   = p.z1[idx];
          const float hn  = p.h1[idx] * z + (1.f - z) * can;
          p.h1[idx]  = hn;
          p.h1b[idx] = (bf16)hn;
        }
      }
    }
    grid_barrier(p.barCnt, p.barGen, nBlocks, gen);
  }

  // ---- Final regressor: out = h1 @ Wreg^T + breg, (64 x 2), fp32, block 0 only.
  if (blockIdx.x == 0) {
    for (int i = threadIdx.x; i < B_ * 2; i += blockDim.x) {
      const int b = i >> 1, o = i & 1;
      float s = p.breg[o];
      const float* hrow = p.h1 + (size_t)b * H_;
      const float* wrow = p.wreg + (size_t)o * H_;
      for (int k = 0; k < H_; ++k) s += hrow[k] * wrow[k];
      p.out[i] = s;
    }
  }
}

// ---------------- fp32 -> bf16 conversion --------------------------------------

__global__ void __launch_bounds__(256)
cvt_f32_bf16(const float* __restrict__ in, bf16* __restrict__ out, size_t n) {
  size_t i = (size_t)blockIdx.x * blockDim.x + threadIdx.x;
  const size_t stride = (size_t)gridDim.x * blockDim.x;
  for (; i < n; i += stride) out[i] = (bf16)in[i];
}

// ---------------- host launcher ------------------------------------------------

extern "C" void kernel_launch(void* const* d_in, const int* in_sizes, int n_in,
                              void* d_out, int out_size, void* d_ws, size_t ws_size,
                              hipStream_t stream) {
  (void)in_sizes; (void)n_in; (void)out_size; (void)ws_size;

  const float* x    = (const float*)d_in[0];
  const float* Wr0  = (const float*)d_in[1];
  const float* br0  = (const float*)d_in[2];
  const float* Wz0  = (const float*)d_in[3];
  const float* bz0  = (const float*)d_in[4];
  const float* Wc0  = (const float*)d_in[5];
  const float* bc0  = (const float*)d_in[6];
  const float* Wr1  = (const float*)d_in[7];
  const float* br1  = (const float*)d_in[8];
  const float* Wz1  = (const float*)d_in[9];
  const float* bz1  = (const float*)d_in[10];
  const float* Wc1  = (const float*)d_in[11];
  const float* bc1  = (const float*)d_in[12];
  const float* Wreg = (const float*)d_in[13];
  const float* breg = (const float*)d_in[14];

  const size_t nW0 = (size_t)H_ * (I_ + H_);   // 1024*1536
  const size_t nW1 = (size_t)H_ * (2 * H_);    // 1024*2048
  const size_t nX  = (size_t)B_ * S_ * I_;
  const size_t nBH = (size_t)B_ * H_;

  char* ws = (char*)d_ws;
  size_t off = 0;
  auto take = [&](size_t bytes) -> char* {
    char* pch = ws + off;
    off = (off + bytes + 255) & ~(size_t)255;
    return pch;
  };

  bf16* wr0b = (bf16*)take(sizeof(bf16) * nW0);
  bf16* wz0b = (bf16*)take(sizeof(bf16) * nW0);
  bf16* wc0b = (bf16*)take(sizeof(bf16) * nW0);
  bf16* wr1b = (bf16*)take(sizeof(bf16) * nW1);
  bf16* wz1b = (bf16*)take(sizeof(bf16) * nW1);
  bf16* wc1b = (bf16*)take(sizeof(bf16) * nW1);
  bf16* xb   = (bf16*)take(sizeof(bf16) * nX);
  float* h0  = (float*)take(sizeof(float) * nBH);
  float* h1  = (float*)take(sizeof(float) * nBH);
  bf16* h0b0 = (bf16*)take(sizeof(bf16) * nBH);
  bf16* h0b1 = (bf16*)take(sizeof(bf16) * nBH);
  bf16* h1b  = (bf16*)take(sizeof(bf16) * nBH);
  bf16* hr0b = (bf16*)take(sizeof(bf16) * nBH);
  bf16* hr1b = (bf16*)take(sizeof(bf16) * nBH);
  float* z0  = (float*)take(sizeof(float) * nBH);
  float* z1  = (float*)take(sizeof(float) * nBH);
  char* barRegion = take(256);
  unsigned* barCnt = (unsigned*)barRegion;
  unsigned* barGen = (unsigned*)(barRegion + 128);

  // Zero hidden state + barrier state (capturable memset nodes).
  hipMemsetAsync(h0,   0, sizeof(float) * nBH, stream);
  hipMemsetAsync(h1,   0, sizeof(float) * nBH, stream);
  hipMemsetAsync(h0b0, 0, sizeof(bf16) * nBH, stream);
  hipMemsetAsync(h0b1, 0, sizeof(bf16) * nBH, stream);
  hipMemsetAsync(h1b,  0, sizeof(bf16) * nBH, stream);
  hipMemsetAsync(barRegion, 0, 256, stream);

  // fp32 -> bf16 conversions (weights stay L2-resident across all 512 steps).
  cvt_f32_bf16<<<1024, 256, 0, stream>>>(Wr0, wr0b, nW0);
  cvt_f32_bf16<<<1024, 256, 0, stream>>>(Wz0, wz0b, nW0);
  cvt_f32_bf16<<<1024, 256, 0, stream>>>(Wc0, wc0b, nW0);
  cvt_f32_bf16<<<1024, 256, 0, stream>>>(Wr1, wr1b, nW1);
  cvt_f32_bf16<<<1024, 256, 0, stream>>>(Wz1, wz1b, nW1);
  cvt_f32_bf16<<<1024, 256, 0, stream>>>(Wc1, wc1b, nW1);
  cvt_f32_bf16<<<2048, 256, 0, stream>>>(x,   xb,   nX);

  GruParams p;
  p.wr0 = wr0b; p.wz0 = wz0b; p.wc0 = wc0b;
  p.wr1 = wr1b; p.wz1 = wz1b; p.wc1 = wc1b;
  p.xb = xb;
  p.h0 = h0; p.h1 = h1;
  p.h0b0 = h0b0; p.h0b1 = h0b1; p.h1b = h1b;
  p.hr0b = hr0b; p.hr1b = hr1b;
  p.z0 = z0; p.z1 = z1;
  p.barCnt = barCnt; p.barGen = barGen;
  p.br0 = br0; p.bz0 = bz0; p.bc0 = bc0;
  p.br1 = br1; p.bz1 = bz1; p.bc1 = bc1;
  p.wreg = Wreg; p.breg = breg;
  p.out = (float*)d_out;

  // 64 blocks x 256 threads = 512 wave32s, co-resident for the grid barrier.
  // G1: 1024 tiles (A||C) -> 2 tiles/wave; G2: 512 tiles (B||D) -> 1 tile/wave.
  gru_persistent<<<64, 256, 0, stream>>>(p);
}
